// InstProtoCLR_46918222741944
// MI455X (gfx1250) — matrix operand order repaired and checked by reading, not compiled, not run on hardware
//
#include <hip/hip_runtime.h>
#include <hip/hip_bf16.h>

#define Bn 4096
#define Dn 1024
#define WAVES 8
#define KSTEPS (Dn / 32)   // 32 k-steps of depth 32

typedef __attribute__((ext_vector_type(16))) _Float16 v16h;
typedef __attribute__((ext_vector_type(8)))  _Float16 v8h;
typedef __attribute__((ext_vector_type(4)))  _Float16 v4h;
typedef __attribute__((ext_vector_type(8)))  float    v8f;

union AFrag { v16h v; v8h p[2]; };
union BFrag { v16h v; v8h p[2]; };

// ---------------------------------------------------------------------------
// Pass 1: normalize anchor rows -> f16 An; gather+normalize centroid rows per
// scale -> f16 Pn_s; invd_s[r] = 1/dens_s[cid].
// grid = (Bn, 4): y==0 -> anchor, y==1..3 -> scale y-1. 256 threads/row.
// ---------------------------------------------------------------------------
__global__ __launch_bounds__(256) void prep_kernel(
    const float* __restrict__ anchor, const int* __restrict__ index,
    const int* __restrict__ i2c0, const float* __restrict__ cent0, const float* __restrict__ dens0,
    const int* __restrict__ i2c1, const float* __restrict__ cent1, const float* __restrict__ dens1,
    const int* __restrict__ i2c2, const float* __restrict__ cent2, const float* __restrict__ dens2,
    _Float16* __restrict__ An, _Float16* __restrict__ PnBase, float* __restrict__ invdBase)
{
    const int row = blockIdx.x;
    const int job = blockIdx.y;
    const int tid = threadIdx.x;

    const float* src;
    _Float16*    dst;
    float*       invd_out = nullptr;
    float        densv    = 1.0f;

    if (job == 0) {
        src = anchor + (size_t)row * Dn;
        dst = An + (size_t)row * Dn;
    } else {
        const int s = job - 1;
        const int*   i2c  = (s == 0) ? i2c0  : (s == 1) ? i2c1  : i2c2;
        const float* cent = (s == 0) ? cent0 : (s == 1) ? cent1 : cent2;
        const float* dens = (s == 0) ? dens0 : (s == 1) ? dens1 : dens2;
        const int cid = i2c[index[row]];
        src      = cent + (size_t)cid * Dn;
        dst      = PnBase + (size_t)s * Bn * Dn + (size_t)row * Dn;
        invd_out = invdBase + (size_t)s * Bn + row;
        densv    = dens[cid];
    }

    // each thread owns 4 consecutive floats (256 * 4 == 1024)
    const float4 v = ((const float4*)src)[tid];
    float ss = v.x * v.x + v.y * v.y + v.z * v.z + v.w * v.w;

    // wave32 butterfly reduce
    #pragma unroll
    for (int m = 16; m >= 1; m >>= 1) ss += __shfl_xor(ss, m, 32);

    __shared__ float red[WAVES];
    const int wid = tid >> 5, lane = tid & 31;
    if (lane == 0) red[wid] = ss;
    __syncthreads();
    if (tid == 0) {
        float t = 0.0f;
        #pragma unroll
        for (int w = 0; w < WAVES; ++w) t += red[w];
        red[0] = t;
    }
    __syncthreads();

    const float nrm = sqrtf(red[0]);
    const float inv = 1.0f / fmaxf(nrm, 1e-12f);

    v4h o;
    o[0] = (_Float16)(v.x * inv);
    o[1] = (_Float16)(v.y * inv);
    o[2] = (_Float16)(v.z * inv);
    o[3] = (_Float16)(v.w * inv);
    ((v4h*)dst)[tid] = o;

    if (tid == 0 && invd_out) *invd_out = 1.0f / densv;
}

// ---------------------------------------------------------------------------
// Lane-local online-logsumexp updates.  Temperature (torch-broadcast over
// logits columns):  c < i : invd[c+1] ;  c == i : invd[0] ;  c > i : invd[c]
// ---------------------------------------------------------------------------

// Wave-uniform case: every element of the tile is on the same side of the
// diagonal -> one scale per lane, branch-free (no compares, no LDS).
__device__ __forceinline__ void tile_stats_uniform(
    const v8f& acc, float sc, float (&M)[8], float (&S)[8])
{
    #pragma unroll
    for (int r = 0; r < 8; ++r) {
        const float v    = acc[r] * sc;
        const float newM = fmaxf(M[r], v);
        S[r] = S[r] * __expf(M[r] - newM) + __expf(v - newM);
        M[r] = newM;
    }
}

// Diagonal sub-tile: full per-element logic + pos-logit capture.
__device__ __forceinline__ void tile_stats_diag(
    const v8f& acc, int c, int ibase, int h,
    float invd0, float invd_c, float invd_c1,
    float (&M)[8], float (&S)[8], float* ldsPos, int i0)
{
    #pragma unroll
    for (int r = 0; r < 8; ++r) {
        const int i = ibase + r + 8 * h;      // C layout: VGPR r -> row r + 8*half
        float v = acc[r];
        const float sc = (c == i) ? invd0 : ((c < i) ? invd_c1 : invd_c);
        v *= sc;
        if (c == i) ldsPos[i - i0] = v;       // unique writer per row in whole block

        const float newM = fmaxf(M[r], v);
        S[r] = S[r] * __expf(M[r] - newM) + __expf(v - newM);
        M[r] = newM;
    }
}

// Pairwise merge of two (M, S) logsumexp states.
__device__ __forceinline__ void merge_ms(float& M, float& S, float oM, float oS)
{
    const float nM = fmaxf(M, oM);
    S = S * __expf(M - nM) + oS * __expf(oM - nM);
    M = nM;
}

// ---------------------------------------------------------------------------
// Pass 2: fused  sim = An @ Pn_s^T  (+ temperature)  -> streaming logsumexp.
// grid = (Bn/32 row tiles, 3 scales), 256 threads = 8 waves.
// 2x2 register blocking: two 16-row A slabs (shared LDS tile) x two 16-col
// B tiles per wave iteration -> 4 accumulators, each B frag reused by 2 WMMAs.
// ---------------------------------------------------------------------------
__global__ __launch_bounds__(256) void sim_ce_kernel(
    const _Float16* __restrict__ An,
    const _Float16* __restrict__ PnBase,
    const float*    __restrict__ invdBase,
    float*          __restrict__ ceBase)
{
    const int s = blockIdx.y;
    const _Float16* Pn   = PnBase   + (size_t)s * Bn * Dn;
    const float*    invd = invdBase + (size_t)s * Bn;
    float*          ce   = ceBase   + (size_t)s * Bn;
    const int i0 = blockIdx.x * 32;

    __shared__ _Float16 ldsA[32 * Dn];      // 64 KB A-tile (f16), rows i0..i0+31
    __shared__ float    ldsM[WAVES][32];
    __shared__ float    ldsS[WAVES][32];
    __shared__ float    ldsPos[32];

    const int tid = threadIdx.x;

    // cooperative A-tile load: rows i0..i0+31 are contiguous (64 KB) in An
    {
        const v8h* srcv = (const v8h*)(An + (size_t)i0 * Dn);
        v8h*       dstv = (v8h*)ldsA;
        #pragma unroll
        for (int q = 0; q < 16; ++q) dstv[q * 256 + tid] = srcv[q * 256 + tid];
    }
    __syncthreads();

    const int wid  = tid >> 5;
    const int lane = tid & 31;
    const int ln   = lane & 15;   // A row / B column / C column within tile
    const int h    = lane >> 4;   // lane-half selects K sub-range / C row half

    const float invd0 = invd[0];

    float runM[2][8], runS[2][8];
    #pragma unroll
    for (int si = 0; si < 2; ++si)
        #pragma unroll
        for (int r = 0; r < 8; ++r) { runM[si][r] = -1e30f; runS[si][r] = 0.0f; }

    // A fragment base: lane holds row m=ln; K = h*16 + {0..15}
    const _Float16* aptr0 = ldsA + (size_t)ln * Dn + h * 16;          // slab 0
    const _Float16* aptr1 = aptr0 + 16 * Dn;                          // slab 1

    for (int jb = wid; jb < Bn / 32; jb += WAVES) {
        const int j0 = jb * 32;

        v8f acc00 = {}, acc10 = {}, acc01 = {}, acc11 = {};
        // B fragment: lane ln holds column n=ln; lanes 0-15 K 0..15, 16-31 K 16..31
        const _Float16* bptr0 = Pn + (size_t)(j0 + ln) * Dn + h * 16; // col tile 0
        const _Float16* bptr1 = bptr0 + 16 * Dn;                      // col tile 1

        #pragma unroll 2
        for (int ks = 0; ks < KSTEPS; ++ks) {
            AFrag a0, a1;
            a0.p[0] = *(const v8h*)(aptr0 + ks * 32);
            a0.p[1] = *(const v8h*)(aptr0 + ks * 32 + 8);
            a1.p[0] = *(const v8h*)(aptr1 + ks * 32);
            a1.p[1] = *(const v8h*)(aptr1 + ks * 32 + 8);
            BFrag b0, b1;
            b0.p[0] = *(const v8h*)(bptr0 + ks * 32);
            b0.p[1] = *(const v8h*)(bptr0 + ks * 32 + 8);
            b1.p[0] = *(const v8h*)(bptr1 + ks * 32);
            b1.p[1] = *(const v8h*)(bptr1 + ks * 32 + 8);

            acc00 = __builtin_amdgcn_wmma_f32_16x16x32_f16(
                false, a0.v, false, b0.v, (short)0, acc00, false, false);
            acc10 = __builtin_amdgcn_wmma_f32_16x16x32_f16(
                false, a1.v, false, b0.v, (short)0, acc10, false, false);
            acc01 = __builtin_amdgcn_wmma_f32_16x16x32_f16(
                false, a0.v, false, b1.v, (short)0, acc01, false, false);
            acc11 = __builtin_amdgcn_wmma_f32_16x16x32_f16(
                false, a1.v, false, b1.v, (short)0, acc11, false, false);
        }

        // Epilogue: classification is wave-uniform except the single diagonal
        // j-block (j0 == i0), which only one wave of one iteration ever sees.
        const int c0 = j0 + ln;
        const int c1 = c0 + 16;
        if (j0 == i0) {
            // acc00 (slab0 x tile0) and acc11 (slab1 x tile1) straddle the
            // diagonal; acc10 is strictly below (c<i), acc01 strictly above.
            const float invd_c0  = invd[c0];
            const float invd_c01 = (c0 + 1 < Bn) ? invd[c0 + 1] : 1.0f;
            const float invd_c1v = invd[c1];
            const float invd_c11 = (c1 + 1 < Bn) ? invd[c1 + 1] : 1.0f;
            tile_stats_diag(acc00, c0, i0,      h, invd0, invd_c0,  invd_c01,
                            runM[0], runS[0], ldsPos, i0);
            tile_stats_uniform(acc10, invd_c01, runM[1], runS[1]);   // c < i
            tile_stats_uniform(acc01, invd_c1v, runM[0], runS[0]);   // c > i
            tile_stats_diag(acc11, c1, i0 + 16, h, invd0, invd_c1v, invd_c11,
                            runM[1], runS[1], ldsPos, i0);
        } else if (j0 < i0) {            // whole j-block below diagonal: c < i
            const float s0 = invd[c0 + 1];
            const float s1 = invd[c1 + 1];
            tile_stats_uniform(acc00, s0, runM[0], runS[0]);
            tile_stats_uniform(acc10, s0, runM[1], runS[1]);
            tile_stats_uniform(acc01, s1, runM[0], runS[0]);
            tile_stats_uniform(acc11, s1, runM[1], runS[1]);
        } else {                          // whole j-block above diagonal: c > i
            const float s0 = invd[c0];
            const float s1 = invd[c1];
            tile_stats_uniform(acc00, s0, runM[0], runS[0]);
            tile_stats_uniform(acc10, s0, runM[1], runS[1]);
            tile_stats_uniform(acc01, s1, runM[0], runS[0]);
            tile_stats_uniform(acc11, s1, runM[1], runS[1]);
        }
    }

    // ONE cross-lane merge at the end: fold the 16 column-slices of each row
    // (butterfly over lanes within each 16-lane half; masks 1,2,4,8).
    #pragma unroll
    for (int si = 0; si < 2; ++si)
        #pragma unroll
        for (int r = 0; r < 8; ++r) {
            float M = runM[si][r], S = runS[si][r];
            #pragma unroll
            for (int mask = 1; mask <= 8; mask <<= 1) {
                const float oM = __shfl_xor(M, mask, 32);
                const float oS = __shfl_xor(S, mask, 32);
                merge_ms(M, S, oM, oS);
            }
            runM[si][r] = M; runS[si][r] = S;
        }

    // publish per-wave stats (one lane per (half, r) is representative)
    if (ln == 0) {
        #pragma unroll
        for (int si = 0; si < 2; ++si)
            #pragma unroll
            for (int r = 0; r < 8; ++r) {
                ldsM[wid][si * 16 + r + 8 * h] = runM[si][r];
                ldsS[wid][si * 16 + r + 8 * h] = runS[si][r];
            }
    }
    __syncthreads();

    if (tid < 32) {
        float gM = -1e30f;
        #pragma unroll
        for (int w = 0; w < WAVES; ++w) gM = fmaxf(gM, ldsM[w][tid]);
        float gS = 0.0f;
        #pragma unroll
        for (int w = 0; w < WAVES; ++w) gS += ldsS[w][tid] * __expf(ldsM[w][tid] - gM);
        const float lse = gM + __logf(gS);
        ce[i0 + tid] = lse - ldsPos[tid];
    }
}

// ---------------------------------------------------------------------------
// Pass 3: loss = ce0/27 + ce1/9 + ce2/3   (division-inside-loop semantics)
// ---------------------------------------------------------------------------
__global__ __launch_bounds__(256) void combine_kernel(
    const float* __restrict__ ceBase, float* __restrict__ loss)
{
    const int i = blockIdx.x * blockDim.x + threadIdx.x;
    if (i < Bn) {
        loss[i] = ceBase[i] * (1.0f / 27.0f)
                + ceBase[Bn + i] * (1.0f / 9.0f)
                + ceBase[2 * Bn + i] * (1.0f / 3.0f);
    }
}

extern "C" void kernel_launch(void* const* d_in, const int* in_sizes, int n_in,
                              void* d_out, int out_size, void* d_ws, size_t ws_size,
                              hipStream_t stream) {
    (void)in_sizes; (void)n_in; (void)out_size; (void)ws_size;

    // setup_inputs() order: anchor, index, (i2c0, cent0, dens0), (i2c1, cent1, dens1), (i2c2, cent2, dens2)
    const float* anchor = (const float*)d_in[0];
    const int*   index  = (const int*)  d_in[1];
    const int*   i2c0   = (const int*)  d_in[2];
    const float* cent0  = (const float*)d_in[3];
    const float* dens0  = (const float*)d_in[4];
    const int*   i2c1   = (const int*)  d_in[5];
    const float* cent1  = (const float*)d_in[6];
    const float* dens1  = (const float*)d_in[7];
    const int*   i2c2   = (const int*)  d_in[8];
    const float* cent2  = (const float*)d_in[9];
    const float* dens2  = (const float*)d_in[10];
    float* loss = (float*)d_out;

    // workspace layout:
    //   An   : Bn*Dn f16           =  8 MB   @ 0
    //   Pn   : 3*Bn*Dn f16         = 24 MB   @ 8 MB
    //   invd : 3*Bn f32            = 48 KB   @ 32 MB
    //   ce   : 3*Bn f32            = 48 KB   @ 32 MB + 48 KB
    char* ws = (char*)d_ws;
    _Float16* An   = (_Float16*)ws;
    _Float16* Pn   = (_Float16*)(ws + ((size_t)8 << 20));
    float*    invd = (float*)(ws + ((size_t)32 << 20));
    float*    ce   = (float*)(ws + ((size_t)32 << 20) + (size_t)3 * Bn * sizeof(float));

    prep_kernel<<<dim3(Bn, 4), 256, 0, stream>>>(
        anchor, index,
        i2c0, cent0, dens0,
        i2c1, cent1, dens1,
        i2c2, cent2, dens2,
        An, Pn, invd);

    sim_ce_kernel<<<dim3(Bn / 32, 3), 256, 0, stream>>>(An, Pn, invd, ce);

    combine_kernel<<<dim3(Bn / 256), 256, 0, stream>>>(ce, loss);
}
